// SingleStreamMultiAttention_55611236549532
// MI455X (gfx1250) — matrix operand order repaired and checked by reading, not compiled, not run on hardware
//
#include <hip/hip_runtime.h>
#include <hip/hip_bf16.h>

// ---------------------------------------------------------------------------
// Problem constants (fixed by the reference setup_inputs)
// ---------------------------------------------------------------------------
constexpr int N_T   = 21;
constexpr int S_SP  = 1560;            // N_h * N_w
constexpr int NTOK  = N_T * S_SP;      // 32760
constexpr int DIM   = 1536;
constexpr int ENC   = 768;
constexpr int NA    = 32;
constexpr int NH    = 12;
constexpr int HD    = 128;
constexpr int BH    = N_T * NH;        // 252

typedef __attribute__((ext_vector_type(16))) __bf16 v16bf;
typedef __attribute__((ext_vector_type(8)))  float  v8f;

#define USE_ASYNC 1
#define LB __launch_bounds__(256, 1)

// ---------------------------------------------------------------------------
// Helpers
// ---------------------------------------------------------------------------
__device__ __forceinline__ unsigned short bftrunc(float f) {
  return (unsigned short)(__float_as_uint(f) >> 16);
}
__device__ __forceinline__ float bf2f(unsigned short h) {
  return __uint_as_float(((unsigned int)h) << 16);
}
// pack two f32 -> packed bf16 pair with one v_perm_b32
__device__ __forceinline__ unsigned int pkbf2(float a, float b) {
  return __builtin_amdgcn_perm(__float_as_uint(b), __float_as_uint(a), 0x07060302u);
}

#if USE_ASYNC
__device__ __forceinline__ void async_ld_b128(const void* gaddr, void* lds_generic) {
  unsigned int l = (unsigned int)(size_t)lds_generic;   // low 32 bits = LDS offset
  asm volatile("global_load_async_to_lds_b128 %0, %1, off"
               :: "v"(l), "v"(gaddr) : "memory");
}
__device__ __forceinline__ void wait_async0() {
  asm volatile("s_wait_asynccnt 0x0" ::: "memory");
}
__device__ __forceinline__ void wait_async6() {
  asm volatile("s_wait_asynccnt 0x6" ::: "memory");
}
#endif

// Load an A fragment (16x32 bf16, MxK) from LDS, row-major [row][32].
__device__ __forceinline__ v16bf ld_frag_a(const unsigned short* sa, int row, int lane) {
  int base = (lane >> 4) * 16;
  const uint4* p0 = (const uint4*)(sa + row * 32 + base);
  const uint4* p1 = (const uint4*)(sa + row * 32 + base + 8);
  union { uint4 q[2]; v16bf v; } u;
  u.q[0] = *p0; u.q[1] = *p1;
  return u.v;
}
// Load a B fragment (32x16 bf16, KxN) from LDS stored as [n][32 K contiguous].
__device__ __forceinline__ v16bf ld_frag_b(const unsigned short* sb, int ncol, int lane) {
  int koff = (lane >> 4) * 16;
  const uint4* p = (const uint4*)(sb + ncol * 32 + koff);
  union { uint4 q[2]; v16bf v; } u;
  u.q[0] = p[0]; u.q[1] = p[1];
  return u.v;
}

// ---------------------------------------------------------------------------
// WMMA GEMM mainloop: C[256x128] = A[256,K] * B[128,K]^T, A/B bf16 row-major.
// A rows [m0, m0+256) must be readable (buffers padded); B rows always valid.
// 8 waves as 4x2; each wave owns a 64x64 sub-tile (4x4 WMMA tiles); K-step 32.
// Double-buffered GLOBAL_LOAD_ASYNC_TO_LDS_B128 pipeline (6 issues/tile).
// smem carve: saA = smem[0..16383] (2 x 256x32), sbB = smem[16384..24575]
// (2 x 128x32). Caller provides >= 24576 ushorts.
// ---------------------------------------------------------------------------
__device__ __forceinline__ void gemm_block_bf16(
    const unsigned short* __restrict__ A, const unsigned short* __restrict__ B,
    int K, int m0, int n0, unsigned short* smem, v8f (&acc)[4][4]) {
  unsigned short* sa = smem;
  unsigned short* sb = smem + 16384;
  const int t = threadIdx.x, lane = t & 31, w = t >> 5;
  const int wm = (w & 3) * 64, wn = (w >> 2) * 64;
  // copy-chunk mapping: chunk c (16B) -> row=c>>2, ko=(c&3)*8
  const int ko = (t & 3) * 8;
  const int ra0 = t >> 2, ra1 = (t + 256) >> 2, ra2 = (t + 512) >> 2, ra3 = (t + 768) >> 2;
  const int rb0 = t >> 2, rb1 = (t + 256) >> 2;

  for (int mt = 0; mt < 4; ++mt)
    for (int nt = 0; nt < 4; ++nt) acc[mt][nt] = 0.0f;

  const int nsteps = K >> 5;
  const unsigned short* Am = A + (size_t)m0 * K;
  const unsigned short* Bn = B + (size_t)n0 * K;

#if USE_ASYNC
  // prologue: tile 0 -> buffer 0
  async_ld_b128(Am + (size_t)ra0 * K + ko, sa + ra0 * 32 + ko);
  async_ld_b128(Am + (size_t)ra1 * K + ko, sa + ra1 * 32 + ko);
  async_ld_b128(Am + (size_t)ra2 * K + ko, sa + ra2 * 32 + ko);
  async_ld_b128(Am + (size_t)ra3 * K + ko, sa + ra3 * 32 + ko);
  async_ld_b128(Bn + (size_t)rb0 * K + ko, sb + rb0 * 32 + ko);
  async_ld_b128(Bn + (size_t)rb1 * K + ko, sb + rb1 * 32 + ko);
  for (int s = 0; s < nsteps; ++s) {
    const int bufA = (s & 1) * 8192;
    const int bufB = (s & 1) * 4096;
    if (s + 1 < nsteps) {
      const int nbA = ((s + 1) & 1) * 8192;
      const int nbB = ((s + 1) & 1) * 4096;
      const int k0 = (s + 1) << 5;
      async_ld_b128(Am + (size_t)ra0 * K + k0 + ko, sa + nbA + ra0 * 32 + ko);
      async_ld_b128(Am + (size_t)ra1 * K + k0 + ko, sa + nbA + ra1 * 32 + ko);
      async_ld_b128(Am + (size_t)ra2 * K + k0 + ko, sa + nbA + ra2 * 32 + ko);
      async_ld_b128(Am + (size_t)ra3 * K + k0 + ko, sa + nbA + ra3 * 32 + ko);
      async_ld_b128(Bn + (size_t)rb0 * K + k0 + ko, sb + nbB + rb0 * 32 + ko);
      async_ld_b128(Bn + (size_t)rb1 * K + k0 + ko, sb + nbB + rb1 * 32 + ko);
      wait_async6();   // current tile's 6 issues complete (in-order)
    } else {
      wait_async0();
    }
    __syncthreads();   // tile visible to all waves
    v16bf af[4], bfr[4];
    for (int mt = 0; mt < 4; ++mt)
      af[mt] = ld_frag_a(sa + bufA, wm + mt * 16 + (lane & 15), lane);
    for (int nt = 0; nt < 4; ++nt)
      bfr[nt] = ld_frag_b(sb + bufB, wn + nt * 16 + (lane & 15), lane);
    for (int mt = 0; mt < 4; ++mt)
      for (int nt = 0; nt < 4; ++nt)
        acc[mt][nt] = __builtin_amdgcn_wmma_f32_16x16x32_bf16(
            false, af[mt], false, bfr[nt], (short)0, acc[mt][nt], false, false);
    __syncthreads();   // reads done before next tile overwrites other buffer
  }
#else
  for (int s = 0; s < nsteps; ++s) {
    const int k0 = s << 5;
    uint4 a0 = *(const uint4*)(Am + (size_t)ra0 * K + k0 + ko);
    uint4 a1 = *(const uint4*)(Am + (size_t)ra1 * K + k0 + ko);
    uint4 a2 = *(const uint4*)(Am + (size_t)ra2 * K + k0 + ko);
    uint4 a3 = *(const uint4*)(Am + (size_t)ra3 * K + k0 + ko);
    uint4 b0 = *(const uint4*)(Bn + (size_t)rb0 * K + k0 + ko);
    uint4 b1 = *(const uint4*)(Bn + (size_t)rb1 * K + k0 + ko);
    __syncthreads();
    *(uint4*)(sa + ra0 * 32 + ko) = a0;
    *(uint4*)(sa + ra1 * 32 + ko) = a1;
    *(uint4*)(sa + ra2 * 32 + ko) = a2;
    *(uint4*)(sa + ra3 * 32 + ko) = a3;
    *(uint4*)(sb + rb0 * 32 + ko) = b0;
    *(uint4*)(sb + rb1 * 32 + ko) = b1;
    __syncthreads();
    v16bf af[4], bfr[4];
    for (int mt = 0; mt < 4; ++mt)
      af[mt] = ld_frag_a(sa, wm + mt * 16 + (lane & 15), lane);
    for (int nt = 0; nt < 4; ++nt)
      bfr[nt] = ld_frag_b(sb, wn + nt * 16 + (lane & 15), lane);
    for (int mt = 0; mt < 4; ++mt)
      for (int nt = 0; nt < 4; ++nt)
        acc[mt][nt] = __builtin_amdgcn_wmma_f32_16x16x32_bf16(
            false, af[mt], false, bfr[nt], (short)0, acc[mt][nt], false, false);
  }
#endif
}

// ---------------------------------------------------------------------------
// K0: fp32 -> bf16 streaming conversion; elements in [n, npad) zero-filled.
// ---------------------------------------------------------------------------
__global__ LB void k_cvt(const float* __restrict__ s,
                         unsigned short* __restrict__ d, long n, long npad) {
  long p = ((long)blockIdx.x * 256 + threadIdx.x) * 8;
  if (p >= npad) return;
  uint4 o;
  if (p < n) {
    const float4* g = (const float4*)(s + p);
    float4 f0 = g[0], f1 = g[1];
    o.x = pkbf2(f0.x, f0.y); o.y = pkbf2(f0.z, f0.w);
    o.z = pkbf2(f1.x, f1.y); o.w = pkbf2(f1.z, f1.w);
  } else {
    o.x = o.y = o.z = o.w = 0u;
  }
  *(uint4*)(d + p) = o;
}

// ---------------------------------------------------------------------------
// K1: min/max of the two attention-map rows
// ---------------------------------------------------------------------------
__global__ LB void k_minmax(const float* __restrict__ m, float* __restrict__ mm) {
  __shared__ float s[4][256];
  float mn0 = 1e30f, mx0 = -1e30f, mn1 = 1e30f, mx1 = -1e30f;
  for (int i = threadIdx.x; i < NTOK; i += 256) {
    float a = m[i], b = m[NTOK + i];
    mn0 = fminf(mn0, a); mx0 = fmaxf(mx0, a);
    mn1 = fminf(mn1, b); mx1 = fmaxf(mx1, b);
  }
  int t = threadIdx.x;
  s[0][t] = mn0; s[1][t] = mx0; s[2][t] = mn1; s[3][t] = mx1;
  __syncthreads();
  for (int stp = 128; stp > 0; stp >>= 1) {
    if (t < stp) {
      s[0][t] = fminf(s[0][t], s[0][t + stp]);
      s[1][t] = fmaxf(s[1][t], s[1][t + stp]);
      s[2][t] = fminf(s[2][t], s[2][t + stp]);
      s[3][t] = fmaxf(s[3][t], s[3][t + stp]);
    }
    __syncthreads();
  }
  if (t == 0) { mm[0] = s[0][0]; mm[1] = s[1][0]; mm[2] = s[2][0]; mm[3] = s[3][0]; }
}

// ---------------------------------------------------------------------------
// K2: per-token rotary position (2-speaker argmax routing)
// ---------------------------------------------------------------------------
__global__ LB void k_pos(const float* __restrict__ m, const float* __restrict__ mm,
                         float* __restrict__ pos) {
  int i = blockIdx.x * 256 + threadIdx.x;
  if (i >= NTOK) return;
  float a = m[i], b = m[NTOK + i];
  float h1 = (a - mm[0]) / (mm[1] - mm[0] + 1e-8f) * 4.0f;          // (0, 4)
  float h2 = (b - mm[2]) / (mm[3] - mm[2] + 1e-8f) * 4.0f + 20.0f;  // (20, 24)
  pos[i] = (a >= b) ? h1 : h2;
}

// ---------------------------------------------------------------------------
// K3: Q projection. Each n-block is exactly one head; epilogue stages the
// full 256x128 bf16 C-tile in LDS, then coalesced 256B-row stores [b,h,s,d].
// grid = (12 n-blocks, 128 m-blocks)
// ---------------------------------------------------------------------------
__global__ LB void k_qproj(const unsigned short* __restrict__ xb,
                           const unsigned short* __restrict__ qwb,
                           const float* __restrict__ qbias,
                           unsigned short* __restrict__ qout) {
  __shared__ unsigned short shmem[32768];   // 64KB: GEMM tiles, then C-tile
  v8f acc[4][4];
  const int n0 = blockIdx.x * 128, m0 = blockIdx.y * 256;
  gemm_block_bf16(xb, qwb, DIM, m0, n0, shmem, acc);
  const int t = threadIdx.x, lane = t & 31, w = t >> 5;
  const int wm = (w & 3) * 64, wn = (w >> 2) * 64;
  for (int mt = 0; mt < 4; ++mt)
    for (int nt = 0; nt < 4; ++nt)
      for (int r = 0; r < 8; ++r) {
        int row = wm + mt * 16 + r + ((lane >> 4) << 3);
        int col = wn + nt * 16 + (lane & 15);
        shmem[row * 128 + col] = bftrunc(acc[mt][nt][r] + qbias[n0 + col]);
      }
  __syncthreads();
  const int h = blockIdx.x;                 // n0 >> 7
  const int b0 = m0 / S_SP, thresh = (b0 + 1) * S_SP;
  const uint4* soq = (const uint4*)shmem;   // 4096 chunks of 8 bf16
  for (int k = 0; k < 16; ++k) {
    int c = t + k * 256;
    int row = c >> 4, part = c & 15;
    int gm = m0 + row;
    if (gm >= NTOK) continue;
    int b = (gm >= thresh) ? (b0 + 1) : b0;
    int s = gm - b * S_SP;
    *(uint4*)(qout + (((size_t)(b * NH + h)) * S_SP + s) * HD + part * 8) = soq[c];
  }
}

// ---------------------------------------------------------------------------
// K4: in-place RoPE on Q (bf16 [bh][s][d]); position per token
// ---------------------------------------------------------------------------
__global__ LB void k_ropeq(unsigned short* __restrict__ q,
                           const float* __restrict__ pos) {
  size_t p = (size_t)blockIdx.x * 256 + threadIdx.x;
  if (p >= (size_t)BH * S_SP * 64) return;
  int d2 = (int)(p & 63);
  size_t rest = p >> 6;                 // bh*1560 + s
  int s  = (int)(rest % S_SP);
  int bh = (int)(rest / S_SP);
  int tok = (bh / NH) * S_SP + s;
  float fr = __expf(-0.14391156831f * (float)d2);   // 10000^(-2*d2/128)
  float ang = pos[tok] * fr;
  float sn, cs; __sincosf(ang, &sn, &cs);
  unsigned int* pp = (unsigned int*)(q + rest * HD + 2 * d2);
  unsigned int pr = *pp;
  float x0 = bf2f((unsigned short)(pr & 0xffff));
  float x1 = bf2f((unsigned short)(pr >> 16));
  *pp = pkbf2(x0 * cs - x1 * sn, x1 * cs + x0 * sn);
}

// ---------------------------------------------------------------------------
// K5: KV projection; K -> fp32 [bh][a][d] (rope next), V -> bf16 [bh][d][a]
// ---------------------------------------------------------------------------
__global__ LB void k_kvproj(const unsigned short* __restrict__ encb,
                            const unsigned short* __restrict__ kvwb,
                            const float* __restrict__ kvb,
                            float* __restrict__ kf,
                            unsigned short* __restrict__ vb) {
  __shared__ unsigned short shmem[32768];
  v8f acc[4][4];
  const int n0 = blockIdx.x * 128, m0 = blockIdx.y * 256;
  gemm_block_bf16(encb, kvwb, ENC, m0, n0, shmem, acc);
  const int lane = threadIdx.x & 31, w = threadIdx.x >> 5;
  const int wm = (w & 3) * 64, wn = (w >> 2) * 64;
  for (int mt = 0; mt < 4; ++mt)
    for (int nt = 0; nt < 4; ++nt)
      for (int r = 0; r < 8; ++r) {
        int row = wm + mt * 16 + r + ((lane >> 4) << 3);
        int col = wn + nt * 16 + (lane & 15);
        int gm = m0 + row;
        if (gm >= N_T * NA) continue;
        int gn = n0 + col;
        float v = acc[mt][nt][r] + kvb[gn];
        int b = gm >> 5, a = gm & 31;
        int c2 = gn >= DIM;
        int rem = c2 ? (gn - DIM) : gn;
        int h = rem >> 7, d = rem & 127;
        int bh = b * NH + h;
        if (!c2)
          kf[(((size_t)bh * NA + a) << 7) + d] = v;
        else
          vb[(((size_t)bh * HD + d) << 5) + a] = bftrunc(v);
      }
}

// ---------------------------------------------------------------------------
// K6: RoPE on K (bucket-centre positions) -> bf16 [bh][a][d]
// ---------------------------------------------------------------------------
__global__ LB void k_ropek(const float* __restrict__ kf,
                           unsigned short* __restrict__ kb) {
  int p = blockIdx.x * 256 + threadIdx.x;
  if (p >= BH * NA * 64) return;
  int d2 = p & 63;
  int rest = p >> 6;                 // bh*32 + a
  int a = rest & 31;
  float posv = (a < 16) ? 2.0f : 22.0f;     // bucket centres
  float fr = __expf(-0.14391156831f * (float)d2);
  float ang = posv * fr;
  float sn, cs; __sincosf(ang, &sn, &cs);
  size_t base = ((size_t)rest << 7) + 2 * d2;
  float x0 = kf[base], x1 = kf[base + 1];
  *(unsigned int*)(kb + base) = pkbf2(x0 * cs - x1 * sn, x1 * cs + x0 * sn);
}

// ---------------------------------------------------------------------------
// K7: fused cross-attention per (b,h): softmax(Q K^T / sqrt(hd)) V
// grid = (13 m-blocks, 252 bh); 8 waves, each owns 16 query rows.
// LDS (32KB): phase1 K[0..4095] V[4096..8191] P[8192..12287]; phase2 C-tile.
// ---------------------------------------------------------------------------
__global__ LB void k_attn(const unsigned short* __restrict__ qb,
                          const unsigned short* __restrict__ kb,
                          const unsigned short* __restrict__ vb,
                          unsigned short* __restrict__ attn) {
  __shared__ unsigned short shmem[16384];
  unsigned short* lds_k = shmem;            // [a][d]
  unsigned short* lds_v = shmem + 4096;     // [d][a]
  unsigned short* lds_p = shmem + 8192;     // per-wave 16x32 probs
  const int t = threadIdx.x, lane = t & 31, w = t >> 5;
  const int bh = blockIdx.y;
  const int b = bh / NH, h = bh % NH;

  {
    const uint4* gk = (const uint4*)(kb + (size_t)bh * (NA * HD));
    const uint4* gv = (const uint4*)(vb + (size_t)bh * (NA * HD));
    uint4* sk = (uint4*)lds_k;
    uint4* sv = (uint4*)lds_v;
    sk[t] = gk[t]; sk[t + 256] = gk[t + 256];
    sv[t] = gv[t]; sv[t + 256] = gv[t + 256];
  }
  __syncthreads();

  const unsigned short* qsl = qb + (size_t)bh * ((size_t)S_SP * HD);
  int row = blockIdx.x * 128 + w * 16 + (lane & 15);
  int rowc = (row < S_SP) ? row : (S_SP - 1);

  // ---- scores: Q[16x128] x K^T -> 16x32 ----
  v8f sacc[2];
  sacc[0] = 0.0f; sacc[1] = 0.0f;
  for (int kk = 0; kk < 4; ++kk) {
    int base = (lane >> 4) * 16;
    const uint4* p0 = (const uint4*)(qsl + (size_t)rowc * HD + kk * 32 + base);
    const uint4* p1 = (const uint4*)(qsl + (size_t)rowc * HD + kk * 32 + base + 8);
    union { uint4 q[2]; v16bf v; } ua;
    ua.q[0] = *p0; ua.q[1] = *p1;
    for (int nt = 0; nt < 2; ++nt) {
      int nl = nt * 16 + (lane & 15);
      int koff = (lane >> 4) * 16;
      const uint4* pb = (const uint4*)(lds_k + nl * HD + kk * 32 + koff);
      union { uint4 q[2]; v16bf v; } ub;
      ub.q[0] = pb[0]; ub.q[1] = pb[1];
      sacc[nt] = __builtin_amdgcn_wmma_f32_16x16x32_bf16(
          false, ua.v, false, ub.v, (short)0, sacc[nt], false, false);
    }
  }

  // ---- softmax over 32 keys per row ----
  const float SC = 0.08838834764831845f;   // 1/sqrt(128)
  for (int r = 0; r < 8; ++r) {
    float x0 = sacc[0][r] * SC;
    float x1 = sacc[1][r] * SC;
    float mx = fmaxf(x0, x1);
    for (int m = 1; m < 16; m <<= 1) mx = fmaxf(mx, __shfl_xor(mx, m));
    float e0 = __expf(x0 - mx), e1 = __expf(x1 - mx);
    float sm = e0 + e1;
    for (int m = 1; m < 16; m <<= 1) sm += __shfl_xor(sm, m);
    float inv = 1.0f / sm;
    int rr = r + ((lane >> 4) << 3);
    int col = lane & 15;
    lds_p[w * 512 + rr * 32 + col]      = bftrunc(e0 * inv);
    lds_p[w * 512 + rr * 32 + col + 16] = bftrunc(e1 * inv);
  }
  __syncthreads();

  // ---- out = P[16x32] x V[32x128] ----
  union { uint4 q[2]; v16bf v; } up;
  {
    int base = (lane >> 4) * 16;
    const uint4* p0 = (const uint4*)(lds_p + w * 512 + (lane & 15) * 32 + base);
    const uint4* p1 = (const uint4*)(lds_p + w * 512 + (lane & 15) * 32 + base + 8);
    up.q[0] = *p0; up.q[1] = *p1;
  }
  v8f oacc[8];
  for (int nt = 0; nt < 8; ++nt) {
    oacc[nt] = 0.0f;
    int nl = nt * 16 + (lane & 15);
    int koff = (lane >> 4) * 16;
    const uint4* pb = (const uint4*)(lds_v + nl * NA + koff);
    union { uint4 q[2]; v16bf v; } ub;
    ub.q[0] = pb[0]; ub.q[1] = pb[1];
    oacc[nt] = __builtin_amdgcn_wmma_f32_16x16x32_bf16(
        false, up.v, false, ub.v, (short)0, oacc[nt], false, false);
  }
  __syncthreads();   // everyone done with K/V/P before reuse

  // ---- stage 128x128 bf16 out-tile, then coalesced 256B-row stores ----
  for (int nt = 0; nt < 8; ++nt)
    for (int r = 0; r < 8; ++r) {
      int lrow = w * 16 + r + ((lane >> 4) << 3);
      int col = nt * 16 + (lane & 15);
      shmem[lrow * 128 + col] = bftrunc(oacc[nt][r]);
    }
  __syncthreads();
  const uint4* soq = (const uint4*)shmem;
  for (int k = 0; k < 8; ++k) {
    int c = t + k * 256;
    int lrow = c >> 4, part = c & 15;
    int srow = blockIdx.x * 128 + lrow;
    if (srow >= S_SP) continue;
    *(uint4*)(attn + ((size_t)(b * S_SP + srow)) * DIM + h * HD + part * 8) = soq[c];
  }
}

// ---------------------------------------------------------------------------
// K8: output projection; fp32 epilogue staged through LDS in two half-tiles.
// ---------------------------------------------------------------------------
__global__ LB void k_oproj(const unsigned short* __restrict__ attn,
                           const unsigned short* __restrict__ pwb,
                           const float* __restrict__ pbias,
                           float* __restrict__ out) {
  __shared__ unsigned short shmem[32768];   // 64KB: GEMM tiles, then f32 half-tile
  v8f acc[4][4];
  const int n0 = blockIdx.x * 128, m0 = blockIdx.y * 256;
  gemm_block_bf16(attn, pwb, DIM, m0, n0, shmem, acc);
  const int t = threadIdx.x, lane = t & 31, w = t >> 5;
  const int wm = (w & 3) * 64, wn = (w >> 2) * 64;
  float* sof = (float*)shmem;               // 256 x 64 f32
  const float4* sofq = (const float4*)shmem;
  for (int half = 0; half < 2; ++half) {
    __syncthreads();
    if (wn == half * 64) {
      for (int mt = 0; mt < 4; ++mt)
        for (int nt = 0; nt < 4; ++nt)
          for (int r = 0; r < 8; ++r) {
            int lrow = wm + mt * 16 + r + ((lane >> 4) << 3);
            int col = nt * 16 + (lane & 15);
            sof[lrow * 64 + col] = acc[mt][nt][r] + pbias[n0 + half * 64 + col];
          }
    }
    __syncthreads();
    for (int k = 0; k < 16; ++k) {
      int c = t + k * 256;                  // 4096 float4 chunks
      int lrow = c >> 4, part = c & 15;
      int gm = m0 + lrow;
      if (gm >= NTOK) continue;
      *(float4*)(out + (size_t)gm * DIM + n0 + half * 64 + part * 4) = sofq[c];
    }
  }
}

// ---------------------------------------------------------------------------
// Launch
// ---------------------------------------------------------------------------
extern "C" void kernel_launch(void* const* d_in, const int* in_sizes, int n_in,
                              void* d_out, int out_size, void* d_ws, size_t ws_size,
                              hipStream_t stream) {
  const float* x    = (const float*)d_in[0];
  const float* enc  = (const float*)d_in[1];
  const float* amap = (const float*)d_in[2];
  const float* qw   = (const float*)d_in[3];
  const float* qb_  = (const float*)d_in[4];
  const float* kvw  = (const float*)d_in[5];
  const float* kvb  = (const float*)d_in[6];
  const float* pw   = (const float*)d_in[7];
  const float* pb   = (const float*)d_in[8];
  float* out = (float*)d_out;

  // workspace layout (all offsets/sizes multiples of 256B)
  char* w = (char*)d_ws;
  const size_t o_mm   = 0;                                   // 4 f32
  const size_t o_pos  = 256;                                 // 32760 f32
  const size_t o_kf   = o_pos  + 131072;                     // 252*32*128 f32
  const size_t o_kb   = o_kf   + 4128768;                    // 252*32*128 bf16
  const size_t o_vb   = o_kb   + 2064384;                    // 252*128*32 bf16
  const size_t o_qb   = o_vb   + 2064384;                    // 252*1560*128 bf16
  const size_t o_attn = o_qb   + 100638720;                  // 32768*1536 bf16 (padded)
  const size_t o_xb   = o_attn + 100663296;                  // 32768*1536 bf16 (padded)
  const size_t o_encb = o_xb   + 100663296;                  // 768*768 bf16 (padded)
  const size_t o_qwb  = o_encb + 1179648;                    // 1536*1536 bf16
  const size_t o_kvwb = o_qwb  + 4718592;                    // 3072*768 bf16
  const size_t o_pwb  = o_kvwb + 4718592;                    // 1536*1536 bf16
  float* mm            = (float*)(w + o_mm);
  float* pos           = (float*)(w + o_pos);
  float* kf            = (float*)(w + o_kf);
  unsigned short* kb   = (unsigned short*)(w + o_kb);
  unsigned short* vb   = (unsigned short*)(w + o_vb);
  unsigned short* qbuf = (unsigned short*)(w + o_qb);
  unsigned short* attn = (unsigned short*)(w + o_attn);
  unsigned short* xb   = (unsigned short*)(w + o_xb);
  unsigned short* encb = (unsigned short*)(w + o_encb);
  unsigned short* qwb  = (unsigned short*)(w + o_qwb);
  unsigned short* kvwb = (unsigned short*)(w + o_kvwb);
  unsigned short* pwb  = (unsigned short*)(w + o_pwb);

  // ---- bf16 pre-conversion (streaming, v_perm packing) ----
  const long n_x   = (long)NTOK * DIM,  np_x   = 32768L * DIM;
  const long n_enc = (long)N_T * NA * ENC, np_enc = 768L * ENC;
  const long n_qw  = (long)DIM * DIM;
  const long n_kvw = (long)2 * DIM * ENC;
  k_cvt<<<(unsigned)(np_x  / 8 / 256), 256, 0, stream>>>(x,   xb,   n_x,   np_x);
  k_cvt<<<(unsigned)(np_enc/ 8 / 256), 256, 0, stream>>>(enc, encb, n_enc, np_enc);
  k_cvt<<<(unsigned)(n_qw  / 8 / 256), 256, 0, stream>>>(qw,  qwb,  n_qw,  n_qw);
  k_cvt<<<(unsigned)(n_kvw / 8 / 256), 256, 0, stream>>>(kvw, kvwb, n_kvw, n_kvw);
  k_cvt<<<(unsigned)(n_qw  / 8 / 256), 256, 0, stream>>>(pw,  pwb,  n_qw,  n_qw);

  // ---- routing positions ----
  k_minmax<<<1, 256, 0, stream>>>(amap, mm);
  k_pos<<<(NTOK + 255) / 256, 256, 0, stream>>>(amap, mm, pos);

  // ---- pipeline ----
  k_qproj<<<dim3(DIM / 128, (NTOK + 255) / 256), 256, 0, stream>>>(xb, qwb, qb_, qbuf);
  k_ropeq<<<(unsigned)(((size_t)BH * S_SP * 64 + 255) / 256), 256, 0, stream>>>(qbuf, pos);
  k_kvproj<<<dim3((2 * DIM) / 128, (N_T * NA + 255) / 256), 256, 0, stream>>>(encb, kvwb, kvb, kf, vb);
  k_ropek<<<(BH * NA * 64 + 255) / 256, 256, 0, stream>>>(kf, kb);
  k_attn<<<dim3((S_SP + 127) / 128, BH), 256, 0, stream>>>(qbuf, kb, vb, attn);
  k_oproj<<<dim3(DIM / 128, (NTOK + 255) / 256), 256, 0, stream>>>(attn, pwb, pb, out);
}
